// TAG_3762391351591
// MI455X (gfx1250) — compile-verified, matrix operand order checked by reading, not run on hardware
//
#include <hip/hip_runtime.h>
#include <hip/hip_bf16.h>

// ---------------------------------------------------------------------------
// MI455X (gfx1250) fused windowed-attention / masked-mixture pipeline.
// bf16 storage, f32 WMMA accumulation (v_wmma_f32_16x16x32_bf16), wave32.
// Round 2: transposed v1 layout (contiguous PV fragments), vectorized LDS
// staging (b64 stores), async global->LDS copy overlapped with S/softmax.
// ---------------------------------------------------------------------------

#define C96  96
#define HW   256
#define TOK  256   // 16x16 window tokens
#define NWIN 2048  // B(8) * 16 * 16 windows

#define USE_ASYNC_LDS_COPY 1   // global_load_async_to_lds_b128 + s_wait_asynccnt

typedef __attribute__((ext_vector_type(16))) __bf16 v16bf;
typedef __attribute__((ext_vector_type(8)))  float  v8f;

union pack4 { __bf16 h[4]; uint2 u; };

__device__ __forceinline__ v8f zero8() {
  v8f z;
#pragma unroll
  for (int i = 0; i < 8; ++i) z[i] = 0.f;
  return z;
}

// Load a 16x32 bf16 fragment (A-layout per CDNA5 ISA 7.12.2; B uses the
// symmetric layout with i = N column, reading B^T-stored data).
//   element e in lane L:  i = i0 + (L&15)
//                         k = k0 + (e&7) + (e>=8 ? 16 : 0) + ((L>=16) ? 8 : 0)
//   value at base[i*si + k*sk]   (sk==1 -> two b128 LDS loads per fragment)
__device__ __forceinline__ v16bf ldfrag(const __bf16* base, int i0, int k0,
                                        int si, int sk) {
  const int lane = threadIdx.x & 31;
  const int i  = i0 + (lane & 15);
  const int kb = k0 + ((lane >> 4) & 1) * 8;
  const __bf16* p = base + i * si + kb * sk;
  v16bf f;
#pragma unroll
  for (int e = 0; e < 8; ++e) {
    f[e]     = p[e * sk];
    f[e + 8] = p[(e + 16) * sk];
  }
  return f;
}

__device__ __forceinline__ v8f wmma32(v16bf a, v16bf b, v8f c) {
  // D = A(16x32 bf16) * B(32x16 bf16) + C(16x16 f32)
  return __builtin_amdgcn_wmma_f32_16x16x32_bf16(false, a, false, b,
                                                 (short)0, c, false, false);
}

// Convert a row-major f32 matrix to bf16 in LDS, 4 elements / thread / iter.
__device__ __forceinline__ void stage_w(__bf16* dst, const float* src,
                                        int n, int tid, int nthr) {
  for (int i = tid * 4; i < n; i += nthr * 4) {
    const float4 wf = *(const float4*)(src + i);
    pack4 p;
    p.h[0] = (__bf16)wf.x; p.h[1] = (__bf16)wf.y;
    p.h[2] = (__bf16)wf.z; p.h[3] = (__bf16)wf.w;
    *(uint2*)(dst + i) = p.u;
  }
}

// ---------------------------------------------------------------------------
// Kernel 1: per-window  v = x @ Wv^T + bv  (WMMA), then t = leaky(v @ Wt^T+bt),
// channel-mean, window variance (ddof=1), binary mask; emit v1 (TRANSPOSED
// window layout [c][t], bf16) and v2 (NCHW, bf16) pre-masked.
// grid = 2048 windows, 256 thr / 8 waves.
// ---------------------------------------------------------------------------
__global__ __launch_bounds__(256) void k1_vproj_mask(
    const float* __restrict__ x, const float* __restrict__ Wv,
    const float* __restrict__ bv, const float* __restrict__ Wt,
    const float* __restrict__ bt, const float* __restrict__ thresh,
    __bf16* __restrict__ v1buf, __bf16* __restrict__ v2img,
    float* __restrict__ maskbuf) {
  extern __shared__ char smem[];
  __bf16* xs   = (__bf16*)(smem);            // [256][96]
  __bf16* wv   = (__bf16*)(smem + 49152);    // [96][96]  == B^T for v-GEMM
  __bf16* vs   = (__bf16*)(smem + 67584);    // [256][96]
  __bf16* wt   = (__bf16*)(smem + 116736);   // [32][96], rows 24..31 zero
  float*  tsum = (float*)(smem + 122880);    // [256] per-token channel-mean
  float*  red  = (float*)(smem + 123904);    // [512]
  float*  bvs  = (float*)(smem + 125952);    // [96]
  float*  bts  = (float*)(smem + 126336);    // [24]
  float*  msh  = (float*)(smem + 126432);    // [1]

  const int tid  = threadIdx.x;
  const int wave = tid >> 5;
  const int lane = tid & 31;
  const int wi   = blockIdx.x;
  const int b    = wi >> 8;
  const int widx = wi & 255;
  const int gy0  = (widx >> 4) * 16, gx0 = (widx & 15) * 16;

  // ---- stage x window (coalesced over gx; packed b64 LDS stores) ----
  {
    const int ty = tid >> 4, tx = tid & 15;
    const float* xp =
        x + (((size_t)b * C96) * HW + (gy0 + ty)) * HW + (gx0 + tx);
    for (int c = 0; c < C96; c += 4) {
      pack4 p;
#pragma unroll
      for (int j = 0; j < 4; ++j)
        p.h[j] = (__bf16)xp[(size_t)(c + j) * HW * HW];
      *(uint2*)(xs + tid * C96 + c) = p.u;
    }
  }
  stage_w(wv, Wv, C96 * C96, tid, 256);
  for (int i = tid * 4; i < 32 * C96; i += 1024) {
    pack4 p;
    if (i < 24 * C96) {
      const float4 wf = *(const float4*)(Wt + i);
      p.h[0] = (__bf16)wf.x; p.h[1] = (__bf16)wf.y;
      p.h[2] = (__bf16)wf.z; p.h[3] = (__bf16)wf.w;
    } else {
      p.h[0] = p.h[1] = p.h[2] = p.h[3] = (__bf16)0.f;
    }
    *(uint2*)(wt + i) = p.u;
  }
  if (tid < C96) bvs[tid] = bv[tid];
  if (tid < 24)  bts[tid] = bt[tid];
  tsum[tid] = 0.f;
  __syncthreads();

  // ---- v = x @ Wv^T : 16 M-tiles / 8 waves, 6 N-tiles, K=96 ----
  v8f accv[2][6];
#pragma unroll
  for (int mt = 0; mt < 2; ++mt)
#pragma unroll
    for (int nt = 0; nt < 6; ++nt) accv[mt][nt] = zero8();
#pragma unroll
  for (int mt = 0; mt < 2; ++mt) {
    const int row0 = (wave * 2 + mt) * 16;
#pragma unroll
    for (int kk = 0; kk < 96; kk += 32) {
      v16bf a = ldfrag(xs, row0, kk, C96, 1);
#pragma unroll
      for (int nt = 0; nt < 6; ++nt)
        accv[mt][nt] = wmma32(a, ldfrag(wv, nt * 16, kk, C96, 1), accv[mt][nt]);
    }
  }
  const int nlo = lane & 15, moff = ((lane >> 4) & 1) * 8;
#pragma unroll
  for (int mt = 0; mt < 2; ++mt)
#pragma unroll
    for (int nt = 0; nt < 6; ++nt)
#pragma unroll
      for (int r = 0; r < 8; ++r) {
        const int m = (wave * 2 + mt) * 16 + r + moff;
        const int n = nt * 16 + nlo;
        vs[m * C96 + n] = (__bf16)(accv[mt][nt][r] + bvs[n]);
      }
  __syncthreads();

  // ---- t = leaky(v @ Wt^T + bt), accumulate channel mean per token ----
  v8f acct[2][2];
#pragma unroll
  for (int mt = 0; mt < 2; ++mt)
#pragma unroll
    for (int nt = 0; nt < 2; ++nt) acct[mt][nt] = zero8();
#pragma unroll
  for (int mt = 0; mt < 2; ++mt) {
    const int row0 = (wave * 2 + mt) * 16;
#pragma unroll
    for (int kk = 0; kk < 96; kk += 32) {
      v16bf a = ldfrag(vs, row0, kk, C96, 1);
#pragma unroll
      for (int nt = 0; nt < 2; ++nt)
        acct[mt][nt] = wmma32(a, ldfrag(wt, nt * 16, kk, C96, 1), acct[mt][nt]);
    }
  }
#pragma unroll
  for (int mt = 0; mt < 2; ++mt)
#pragma unroll
    for (int nt = 0; nt < 2; ++nt)
#pragma unroll
      for (int r = 0; r < 8; ++r) {
        const int n = nt * 16 + nlo;
        if (n < 24) {
          const int m = (wave * 2 + mt) * 16 + r + moff;
          float tv = acct[mt][nt][r] + bts[n];
          tv = (tv >= 0.f) ? tv : 0.1f * tv;
          atomicAdd(&tsum[m], tv * (1.0f / 24.0f));   // ds_add_f32
        }
      }
  __syncthreads();

  // ---- variance over 256 tokens (ddof=1) -> binary mask ----
  {
    const float tb = tsum[tid];
    red[tid] = tb;
    red[256 + tid] = tb * tb;
  }
  __syncthreads();
  for (int s2 = 128; s2 > 0; s2 >>= 1) {
    if (tid < s2) {
      red[tid] += red[tid + s2];
      red[256 + tid] += red[256 + tid + s2];
    }
    __syncthreads();
  }
  if (tid == 0) {
    const float sum = red[0], sq = red[256];
    const float var = (sq - sum * sum * (1.0f / 256.0f)) * (1.0f / 255.0f);
    // soft>0.5  <=>  sigmoid(S*(thresh-var))<0.5  <=>  var>thresh
    const float m = (var > thresh[0]) ? 1.0f : 0.0f;
    msh[0] = m;
    maskbuf[wi] = m;
  }
  __syncthreads();
  const float m = msh[0];

  // ---- emit v1 = v*m TRANSPOSED [c][t] (coalesced t-major writes) ----
  {
    __bf16* v1p = v1buf + (size_t)wi * TOK * C96;
    for (int i = tid; i < TOK * C96; i += 256) {
      const int c = i >> 8, t = i & 255;
      v1p[i] = (m != 0.f) ? vs[t * C96 + c] : (__bf16)0.f;
    }
  }
  // ---- emit v2 = v*(1-m) NCHW ----
  {
    const int ty = tid >> 4, tx = tid & 15;
    __bf16* v2p =
        v2img + (((size_t)b * C96) * HW + (gy0 + ty)) * HW + (gx0 + tx);
    for (int c = 0; c < C96; ++c)
      v2p[(size_t)c * HW * HW] = (m != 0.f) ? (__bf16)0.f : vs[tid * C96 + c];
  }
}

// ---------------------------------------------------------------------------
// Kernel 2: per-window attention.  m==0 windows: f = attn @ 0 = 0 (skip).
// 16 waves; wave w owns S-strip rows [16w,16w+16): S strip = 16 N-tile f32
// accumulators (128 VGPRs), softmax via 16-lane shfl reductions, P stored
// bf16 in LDS (128 KB), then f = P @ v1 with WMMA.  The v1 tile is copied
// global->LDS with ASYNC loads overlapped with the S/softmax phase.
// grid = 2048, 512 thr / 16 waves.
// ---------------------------------------------------------------------------
__global__ __launch_bounds__(512) void k2_attn(
    const float* __restrict__ x, const float* __restrict__ Wq,
    const float* __restrict__ bq, const float* __restrict__ Wk,
    const float* __restrict__ bk, const __bf16* __restrict__ v1buf,
    const float* __restrict__ maskbuf, __bf16* __restrict__ fbuf) {
  extern __shared__ char smem[];
  __bf16* qs  = (__bf16*)(smem);             // [256][96]
  __bf16* ks  = (__bf16*)(smem + 49152);     // [256][96]
  __bf16* vs  = (__bf16*)(smem + 98304);     // [96][256] transposed (ph1: xs)
  __bf16* xs  = vs;
  __bf16* wqs = (__bf16*)(smem + 147456);    // [96][96]   (overlaid by Ps)
  __bf16* wks = (__bf16*)(smem + 165888);    // [96][96]   (overlaid by Ps)
  __bf16* Ps  = (__bf16*)(smem + 147456);    // [256][256] bf16 = 128 KB
  float*  bqs = (float*)(smem + 278528);     // [96]
  float*  bks = (float*)(smem + 278912);     // [96]
  const unsigned kVsLdsByteOff = 98304u;     // static offset of vs in LDS

  const int tid = threadIdx.x, wave = tid >> 5, lane = tid & 31;
  const int wi = blockIdx.x;
  __bf16* fp = fbuf + (size_t)wi * TOK * C96;

  const float m = maskbuf[wi];
  if (m == 0.f) {  // pruned window: attention output is exactly zero
    for (int j = tid; j < TOK * C96 / 8; j += 512)
      ((uint4*)fp)[j] = make_uint4(0u, 0u, 0u, 0u);
    return;
  }

  const int b = wi >> 8, widx = wi & 255;
  const int gy0 = (widx >> 4) * 16, gx0 = (widx & 15) * 16;

  // ---- stage x window, Wq, Wk, biases ----
  {
    const int half = tid >> 8;           // two 48-channel halves
    const int t = tid & 255;
    const int ty = t >> 4, tx = t & 15;
    const float* xp =
        x + (((size_t)b * C96 + half * 48) * HW + (gy0 + ty)) * HW + (gx0 + tx);
    for (int c = 0; c < 48; c += 4) {
      pack4 p;
#pragma unroll
      for (int j = 0; j < 4; ++j)
        p.h[j] = (__bf16)xp[(size_t)(c + j) * HW * HW];
      *(uint2*)(xs + t * C96 + half * 48 + c) = p.u;
    }
  }
  stage_w(wqs, Wq, C96 * C96, tid, 512);
  stage_w(wks, Wk, C96 * C96, tid, 512);
  if (tid < C96) {
    bqs[tid] = bq[tid];
    bks[tid] = bk[tid];
  }
  __syncthreads();

  const int nlo = lane & 15, moff = ((lane >> 4) & 1) * 8;

  // ---- q = x@Wq^T+bq, k = x@Wk^T+bk : wave w -> M-tile w ----
  {
    v8f aq[6], ak[6];
#pragma unroll
    for (int nt = 0; nt < 6; ++nt) { aq[nt] = zero8(); ak[nt] = zero8(); }
    const int row0 = wave * 16;
#pragma unroll
    for (int kk = 0; kk < 96; kk += 32) {
      v16bf a = ldfrag(xs, row0, kk, C96, 1);
#pragma unroll
      for (int nt = 0; nt < 6; ++nt) {
        aq[nt] = wmma32(a, ldfrag(wqs, nt * 16, kk, C96, 1), aq[nt]);
        ak[nt] = wmma32(a, ldfrag(wks, nt * 16, kk, C96, 1), ak[nt]);
      }
    }
#pragma unroll
    for (int nt = 0; nt < 6; ++nt)
#pragma unroll
      for (int r = 0; r < 8; ++r) {
        const int mm = row0 + r + moff;
        const int n = nt * 16 + nlo;
        qs[mm * C96 + n] = (__bf16)(aq[nt][r] + bqs[n]);
        ks[mm * C96 + n] = (__bf16)(ak[nt][r] + bks[n]);
      }
  }
  __syncthreads();  // qs/ks complete; xs/wqs/wks regions now reusable

  // ---- launch async global->LDS copy of v1 tile (runs during S/softmax) ----
  {
    const __bf16* gsrc = v1buf + (size_t)wi * TOK * C96;
#if USE_ASYNC_LDS_COPY
    for (int j = tid; j < TOK * C96 / 8; j += 512) {
      const unsigned lds_byte = kVsLdsByteOff + (unsigned)j * 16u;
      const unsigned long long ga =
          (unsigned long long)(const void*)(gsrc + j * 8);
      asm volatile("global_load_async_to_lds_b128 %0, %1, off"
                   :: "v"(lds_byte), "v"(ga) : "memory");
    }
#else
    for (int j = tid; j < TOK * C96 / 8; j += 512)
      ((uint4*)vs)[j] = ((const uint4*)gsrc)[j];
#endif
  }

  // ---- S strip = q(strip) @ k^T : 16 N-tiles, K=96 ----
  v8f s[16];
#pragma unroll
  for (int nt = 0; nt < 16; ++nt) s[nt] = zero8();
  {
    v16bf aq3[3];
#pragma unroll
    for (int kk3 = 0; kk3 < 3; ++kk3)
      aq3[kk3] = ldfrag(qs, wave * 16, kk3 * 32, C96, 1);
#pragma unroll
    for (int nt = 0; nt < 16; ++nt)
#pragma unroll
      for (int kk3 = 0; kk3 < 3; ++kk3)
        s[nt] = wmma32(aq3[kk3], ldfrag(ks, nt * 16, kk3 * 32, C96, 1), s[nt]);
  }

  // ---- row softmax (rows live in 16-lane halves; shfl-xor reductions) ----
#pragma unroll
  for (int r = 0; r < 8; ++r) {
    float mx = -3.4e38f;
#pragma unroll
    for (int nt = 0; nt < 16; ++nt) mx = fmaxf(mx, s[nt][r]);
#pragma unroll
    for (int d = 1; d < 16; d <<= 1) mx = fmaxf(mx, __shfl_xor(mx, d));
    float sum = 0.f;
#pragma unroll
    for (int nt = 0; nt < 16; ++nt) {
      const float e = __expf(s[nt][r] - mx);
      s[nt][r] = e;
      sum += e;
    }
#pragma unroll
    for (int d = 1; d < 16; d <<= 1) sum += __shfl_xor(sum, d);
    const float inv = 1.0f / sum;
    const int row = wave * 16 + r + moff;
#pragma unroll
    for (int nt = 0; nt < 16; ++nt)
      Ps[row * 256 + nt * 16 + nlo] = (__bf16)(s[nt][r] * inv);
  }
#if USE_ASYNC_LDS_COPY
  asm volatile("s_wait_asynccnt 0x0" ::: "memory");  // v1 tile landed in LDS
#endif
  __syncthreads();  // Ps and vs complete

  // ---- f(strip) = P(strip) @ v1 : 6 channel tiles, K=256 ----
  // vs is [c][t]  =>  B fragments are contiguous-K (two b128 loads each)
  {
    v16bf pa[8];
#pragma unroll
    for (int kt = 0; kt < 8; ++kt)
      pa[kt] = ldfrag(Ps, wave * 16, kt * 32, 256, 1);
#pragma unroll
    for (int cn = 0; cn < 6; ++cn) {
      v8f f = zero8();
#pragma unroll
      for (int kt = 0; kt < 8; ++kt)
        f = wmma32(pa[kt], ldfrag(vs, cn * 16, kt * 32, TOK, 1), f);
#pragma unroll
      for (int r = 0; r < 8; ++r) {
        const int row = wave * 16 + r + moff;
        fp[row * C96 + cn * 16 + nlo] = (__bf16)f[r];
      }
    }
  }
}

// ---------------------------------------------------------------------------
// Kernel 3: depthwise 3x3 conv (dilation 1 or 2, same-pad), bf16 in/out,
// f32 accumulate.  Pure streaming kernel (~400 MB traffic for both passes).
// ---------------------------------------------------------------------------
__global__ __launch_bounds__(256) void k3_dwconv(
    const __bf16* __restrict__ in, const float* __restrict__ Wd,
    const float* __restrict__ bd, __bf16* __restrict__ out, int dil) {
  const size_t idx = (size_t)blockIdx.x * 256 + threadIdx.x;
  const int xx = (int)(idx & 255);
  const int yy = (int)((idx >> 8) & 255);
  const int bc = (int)(idx >> 16);
  const int c  = bc % C96;
  const __bf16* ip = in + (idx - (size_t)xx - ((size_t)yy << 8));  // plane base
  const float* w = Wd + c * 9;
  __builtin_prefetch(ip + ((size_t)yy << 8), 0, 1);  // global_prefetch_b8
  float acc = bd[c];
#pragma unroll
  for (int ky = 0; ky < 3; ++ky) {
    const int sy = yy + (ky - 1) * dil;
    if (sy < 0 || sy > 255) continue;
#pragma unroll
    for (int kx = 0; kx < 3; ++kx) {
      const int sx = xx + (kx - 1) * dil;
      if (sx < 0 || sx > 255) continue;
      acc += w[ky * 3 + kx] * (float)ip[(sy << 8) + sx];
    }
  }
  out[idx] = (__bf16)acc;
}

// ---------------------------------------------------------------------------
// Kernel 4: out = (f + v2c) @ Wo^T + bo, per window, f32 NCHW output.
// grid = 2048, 256 thr / 8 waves.
// ---------------------------------------------------------------------------
__global__ __launch_bounds__(256) void k4_outproj(
    const __bf16* __restrict__ fbuf, const __bf16* __restrict__ v2c,
    const float* __restrict__ Wo, const float* __restrict__ bo,
    float* __restrict__ outp) {
  extern __shared__ char smem[];
  __bf16* gs  = (__bf16*)(smem);           // [256][96]
  __bf16* wos = (__bf16*)(smem + 49152);   // [96][96]
  float*  bos = (float*)(smem + 67584);    // [96]

  const int tid = threadIdx.x, wave = tid >> 5, lane = tid & 31;
  const int wi = blockIdx.x;
  const int b = wi >> 8, widx = wi & 255;
  const int gy0 = (widx >> 4) * 16, gx0 = (widx & 15) * 16;

  {
    const int ty = tid >> 4, tx = tid & 15;
    const __bf16* vp =
        v2c + (((size_t)b * C96) * HW + (gy0 + ty)) * HW + (gx0 + tx);
    const __bf16* fp = fbuf + (size_t)wi * TOK * C96 + tid * C96;
    for (int c = 0; c < C96; c += 4) {
      pack4 fl, p;
      fl.u = *(const uint2*)(fp + c);
#pragma unroll
      for (int j = 0; j < 4; ++j)
        p.h[j] = (__bf16)((float)fl.h[j] +
                          (float)vp[(size_t)(c + j) * HW * HW]);
      *(uint2*)(gs + tid * C96 + c) = p.u;
    }
  }
  stage_w(wos, Wo, C96 * C96, tid, 256);
  if (tid < C96) bos[tid] = bo[tid];
  __syncthreads();

  v8f acc[2][6];
#pragma unroll
  for (int mt = 0; mt < 2; ++mt)
#pragma unroll
    for (int nt = 0; nt < 6; ++nt) acc[mt][nt] = zero8();
#pragma unroll
  for (int mt = 0; mt < 2; ++mt) {
    const int row0 = (wave * 2 + mt) * 16;
#pragma unroll
    for (int kk = 0; kk < 96; kk += 32) {
      v16bf a = ldfrag(gs, row0, kk, C96, 1);
#pragma unroll
      for (int nt = 0; nt < 6; ++nt)
        acc[mt][nt] = wmma32(a, ldfrag(wos, nt * 16, kk, C96, 1), acc[mt][nt]);
    }
  }
  const int nlo = lane & 15, moff = ((lane >> 4) & 1) * 8;
#pragma unroll
  for (int mt = 0; mt < 2; ++mt)
#pragma unroll
    for (int nt = 0; nt < 6; ++nt)
#pragma unroll
      for (int r = 0; r < 8; ++r) {
        const int t = (wave * 2 + mt) * 16 + r + moff;
        const int c = nt * 16 + nlo;
        const int gy = gy0 + (t >> 4), gx = gx0 + (t & 15);
        outp[(((size_t)b * C96 + c) * HW + gy) * HW + gx] =
            acc[mt][nt][r] + bos[c];
      }
}

// ---------------------------------------------------------------------------
extern "C" void kernel_launch(void* const* d_in, const int* in_sizes, int n_in,
                              void* d_out, int out_size, void* d_ws,
                              size_t ws_size, hipStream_t stream) {
  (void)in_sizes; (void)n_in; (void)out_size; (void)ws_size;
  const float* x      = (const float*)d_in[0];
  const float* Wv     = (const float*)d_in[1];
  const float* bv     = (const float*)d_in[2];
  const float* Wt     = (const float*)d_in[3];
  const float* bt     = (const float*)d_in[4];
  const float* thresh = (const float*)d_in[5];
  const float* Wq     = (const float*)d_in[6];
  const float* bq     = (const float*)d_in[7];
  const float* Wk     = (const float*)d_in[8];
  const float* bk     = (const float*)d_in[9];
  const float* Wd1    = (const float*)d_in[10];
  const float* bd1    = (const float*)d_in[11];
  const float* Wd2    = (const float*)d_in[12];
  const float* bd2    = (const float*)d_in[13];
  const float* Wo     = (const float*)d_in[14];
  const float* bo     = (const float*)d_in[15];

  constexpr size_t VB = (size_t)8 * C96 * HW * HW * sizeof(__bf16);  // 96 MiB
  char* ws = (char*)d_ws;
  float*  maskbuf = (float*)ws;                       // 2048 f32
  __bf16* v1buf   = (__bf16*)(ws + 8192);             // [wi][c][t] transposed
  __bf16* v2a     = (__bf16*)(ws + 8192 + VB);        // NCHW
  __bf16* v2b     = (__bf16*)(ws + 8192 + 2 * VB);    // NCHW
  __bf16* fbuf    = (__bf16*)(ws + 8192 + 3 * VB);    // [wi][t][c]

  k1_vproj_mask<<<NWIN, 256, 126464, stream>>>(x, Wv, bv, Wt, bt, thresh,
                                               v1buf, v2a, maskbuf);
  k2_attn<<<NWIN, 512, 279296, stream>>>(x, Wq, bq, Wk, bk, v1buf, maskbuf,
                                         fbuf);
  k3_dwconv<<<196608, 256, 0, stream>>>(v2a, Wd1, bd1, v2b, 1);
  k3_dwconv<<<196608, 256, 0, stream>>>(v2b, Wd2, bd2, v2a, 2);
  k4_outproj<<<NWIN, 256, 67968, stream>>>(fbuf, v2a, Wo, bo, (float*)d_out);
}
